// ConversationLSTM_4312147165318
// MI455X (gfx1250) — matrix operand order repaired and checked by reading, not compiled, not run on hardware
//
#include <hip/hip_runtime.h>
#include <stdint.h>

// ---------- types for WMMA ----------
typedef __attribute__((ext_vector_type(16))) __bf16 v16bf;
typedef __attribute__((ext_vector_type(8)))  float  v8f;

union Frag { uint4 u[2]; v16bf v; };

// ---------- problem dims ----------
#define TT 128      // timesteps
#define BB 256      // batch per text
#define RR 768      // 3 * BB fused batch
#define HH 512      // hidden
#define KK 1024     // 2H (concat input)
#define ZZ 2048     // 4H (gates)
#define NWG 32      // one workgroup per 16-wide h-column tile
#define THREADS 256
#define LDSTR 1032  // padded LDS row stride (ushort elems): 516 dwords ≡ 4 (mod 64 banks)

__device__ __forceinline__ unsigned short f2bf(float f) {
  unsigned int u = __float_as_uint(f);
  u += 0x7FFFu + ((u >> 16) & 1u);        // round-to-nearest-even
  return (unsigned short)(u >> 16);
}
__device__ __forceinline__ float bf2f(unsigned short h) {
  return __uint_as_float(((unsigned int)h) << 16);
}

// branchless activations on the recurrent critical path (no exec-mask branches)
__device__ __forceinline__ float fast_sigmoid(float x) {
  return 1.f / (1.f + __expf(-x));
}
__device__ __forceinline__ float fast_tanh(float x) {
  // tanh(x) = 1 - 2/(e^{2x}+1); saturates correctly for |x| large
  return 1.f - 2.f / (__expf(2.f * x) + 1.f);
}

// ---------- weight pre-cast kernels ----------
__global__ void cast_lstm_w_kernel(const float* __restrict__ W,        // [3][1024][2048]
                                   unsigned short* __restrict__ WT) {  // [3][2048][1024]
  int idx = blockIdx.x * blockDim.x + threadIdx.x;
  const int total = 3 * ZZ * KK;
  if (idx >= total) return;
  int k = idx % KK;
  int n = (idx / KK) % ZZ;
  int l = idx / (KK * ZZ);
  WT[idx] = f2bf(W[((size_t)l * KK + k) * ZZ + n]);
}

__global__ void cast_w1_kernel(const float* __restrict__ W1,           // [1536][1024]
                               unsigned short* __restrict__ W1T) {     // [1024][1536]
  int idx = blockIdx.x * blockDim.x + threadIdx.x;
  if (idx >= 1024 * 1536) return;
  int k = idx % 1536;
  int n = idx / 1536;
  W1T[idx] = f2bf(W1[(size_t)k * 1024 + n]);
}

__global__ void cast_w2_kernel(const float* __restrict__ W2,           // [1024][102]
                               const float* __restrict__ b2,
                               unsigned short* __restrict__ W2T,       // [112][1024] padded
                               float* __restrict__ b2p) {              // [112]
  int idx = blockIdx.x * blockDim.x + threadIdx.x;
  if (idx >= 112 * 1024) return;
  int k = idx % 1024;
  int n = idx / 1024;
  W2T[idx] = (n < 102) ? f2bf(W2[(size_t)k * 102 + n]) : (unsigned short)0;
  if (idx < 112) b2p[idx] = (idx < 102) ? b2[idx] : 0.f;
}

// ---------- embedding gather (3 texts fused, bf16) ----------
__global__ void embed_gather_kernel(const int* __restrict__ t1, const int* __restrict__ t2,
                                    const int* __restrict__ t3,
                                    const float* __restrict__ emb,      // [V][512]
                                    unsigned short* __restrict__ X) {   // [T][768][512]
  int idx = blockIdx.x * blockDim.x + threadIdx.x;     // one thread per 8 elements
  const int total = TT * RR * (HH / 8);
  if (idx >= total) return;
  int h8 = idx & (HH / 8 - 1);
  int r  = (idx / (HH / 8)) % RR;
  int t  = idx / ((HH / 8) * RR);
  int text = r >> 8;
  int b = r & 255;
  const int* tok = (text == 0) ? t1 : ((text == 1) ? t2 : t3);
  int w = tok[b * TT + t];
  const float4* s = (const float4*)(emb + (size_t)w * HH + h8 * 8);
  float4 lo = s[0], hi = s[1];
  uint4 pk;
  pk.x = (unsigned int)f2bf(lo.x) | ((unsigned int)f2bf(lo.y) << 16);
  pk.y = (unsigned int)f2bf(lo.z) | ((unsigned int)f2bf(lo.w) << 16);
  pk.z = (unsigned int)f2bf(hi.x) | ((unsigned int)f2bf(hi.y) << 16);
  pk.w = (unsigned int)f2bf(hi.z) | ((unsigned int)f2bf(hi.w) << 16);
  *(uint4*)(X + ((size_t)t * RR + r) * HH + h8 * 8) = pk;
}

// ---------- grid barrier (monotonic counter; reset before each launch) ----------
__global__ void reset_barrier_kernel(int* cnt) { if (threadIdx.x == 0) *cnt = 0; }

__device__ __forceinline__ void grid_sync(int* cnt, int serial) {
  __threadfence();
  __syncthreads();
  if (threadIdx.x == 0) {
    atomicAdd(cnt, 1);
    volatile int* v = cnt;
    int target = serial * NWG;
    while (*v < target) { __builtin_amdgcn_s_sleep(8); }
  }
  __syncthreads();
  __threadfence();
}

// ---------- persistent LSTM layer ----------
// z = [x_t, h] @ W + b ; gate order i, j, f, o (each HH wide).
// Workgroup wg owns h-column tile hc = wg (cols [hc*16, hc*16+16)).
// Its weight slice (4 gates x 16 cols x 1024 K bf16 = 128 KB) is staged into
// LDS ONCE per layer and re-read via ds_load_b128 for all 128 timesteps.
// Each of the 8 waves owns 6 row-tiles; per quad the 4 gate tiles are computed
// together so the gate math + c/h update happen entirely in registers.
__global__ void __launch_bounds__(THREADS)
lstm_layer_kernel(const unsigned short* __restrict__ WT,   // [2048][1024] bf16 (transposed)
                  const float* __restrict__ bias,          // [2048]
                  const unsigned short* __restrict__ Xin,  // [T][768][512] bf16
                  unsigned short* __restrict__ Xout,       // [T][768][512] bf16
                  unsigned short* __restrict__ hA,
                  unsigned short* __restrict__ hB,
                  float* __restrict__ cst,                 // [768][512]
                  int* cnt) {
  extern __shared__ char lds_raw[];
  unsigned short* ldsW = (unsigned short*)lds_raw;         // [64][LDSTR]

  const int hc   = blockIdx.x;                             // h-column tile [0,32)
  const int wv   = threadIdx.x >> 5;                       // wave in wg [0,8)
  const int lane = threadIdx.x & 31;
  const int half = lane >> 4;
  const int lcol = lane & 15;
  const int col  = hc * 16 + lcol;                         // this lane's h column
  int serial = 0;

  // ---- stage this workgroup's weight slice into LDS (once per layer) ----
  for (int it = threadIdx.x; it < 64 * (KK / 8); it += THREADS) {
    int row = it >> 7;                  // [0,64): g*16 + c
    int k8  = it & 127;                 // uint4 index within the K dimension
    int g = row >> 4, c = row & 15;
    int ng = g * HH + hc * 16 + c;      // global weight row
    *(uint4*)(ldsW + row * LDSTR + k8 * 8) =
        *(const uint4*)(WT + (size_t)ng * KK + k8 * 8);
  }

  // ---- zero initial h (hA) and c for this workgroup's columns ----
  for (int i = 0; i < 6; ++i) {
    int rt = wv * 6 + i;
    for (int v = 0; v < 8; ++v) {
      int row = rt * 16 + v + 8 * half;
      hA[(size_t)row * HH + col] = 0;
      cst[(size_t)row * HH + col] = 0.f;
    }
  }

  // per-lane gate biases are loop-invariant
  const float bi = bias[0 * HH + col];
  const float bj = bias[1 * HH + col];
  const float bf = bias[2 * HH + col];
  const float bo = bias[3 * HH + col];

  grid_sync(cnt, ++serial);            // also orders the LDS staging (__syncthreads inside)

  for (int t = 0; t < TT; ++t) {
    const unsigned short* Xt = Xin + (size_t)t * RR * HH;
    unsigned short* Yt = Xout + (size_t)t * RR * HH;
    const unsigned short* hr = (t & 1) ? hB : hA;
    unsigned short* hw = (t & 1) ? hA : hB;

    for (int i = 0; i < 6; ++i) {
      const int rt = wv * 6 + i;
      v8f acc[4] = {};
      const int arow = rt * 16 + lcol;

      for (int kb = 0; kb < 32; ++kb) {
        int ks = (kb & 15) * 32;
        const unsigned short* src = (kb < 16) ? Xt : hr;
        const unsigned short* ap = src + (size_t)arow * HH + ks + half * 8;
        __builtin_prefetch(ap + 32, 0, 3);
        const uint4* pa = (const uint4*)ap;
        Frag a; a.u[0] = pa[0]; a.u[1] = pa[2];
#pragma unroll
        for (int g = 0; g < 4; ++g) {
          const uint4* pb =
              (const uint4*)(ldsW + (g * 16 + lcol) * LDSTR + kb * 32 + half * 8);
          Frag b; b.u[0] = pb[0]; b.u[1] = pb[2];
          acc[g] = __builtin_amdgcn_wmma_f32_16x16x32_bf16(
              false, a.v, false, b.v, (short)0, acc[g], false, false);
        }
      }

      // gates + state update (branchless activations)
#pragma unroll
      for (int v = 0; v < 8; ++v) {
        int row = rt * 16 + v + 8 * half;
        float zi = acc[0][v] + bi;
        float zj = acc[1][v] + bj;
        float zf = acc[2][v] + bf;
        float zo = acc[3][v] + bo;
        float si = fast_sigmoid(zi);
        float sf = fast_sigmoid(zf);
        float so = fast_sigmoid(zo);
        float cn = sf * cst[(size_t)row * HH + col] + si * fast_tanh(zj);
        float hn = so * fast_tanh(cn);
        cst[(size_t)row * HH + col] = cn;
        unsigned short hb = f2bf(hn);
        hw[(size_t)row * HH + col] = hb;
        Yt[(size_t)row * HH + col] = hb;
      }
    }
    grid_sync(cnt, ++serial);
  }
}

// ---------- generic WMMA GEMM: C[M][N] = A[M][K](bf16) @ BT[N][K](bf16) + bias; act 1 = SELU
__global__ void __launch_bounds__(THREADS)
wmma_gemm_kernel(const unsigned short* __restrict__ A,
                 const unsigned short* __restrict__ BT,
                 const float* __restrict__ bias,
                 float* __restrict__ C,
                 int M, int N, int K, int act) {
  const int wave = (blockIdx.x * blockDim.x + (int)threadIdx.x) >> 5;
  const int lane = threadIdx.x & 31;
  const int half = lane >> 4;
  const int lcol = lane & 15;
  const int nt = N >> 4;
  const int ntiles = (M >> 4) * nt;
  const int nwaves = (gridDim.x * blockDim.x) >> 5;

  for (int tile = wave; tile < ntiles; tile += nwaves) {
    int rt = tile / nt, ct = tile % nt;
    v8f acc = {};
    const int arow = rt * 16 + lcol;
    const int brow = ct * 16 + lcol;
    for (int kb = 0; kb < (K >> 5); ++kb) {
      const uint4* pa = (const uint4*)(A + (size_t)arow * K + kb * 32 + half * 8);
      Frag a; a.u[0] = pa[0]; a.u[1] = pa[2];
      const uint4* pb = (const uint4*)(BT + (size_t)brow * K + kb * 32 + half * 8);
      Frag b; b.u[0] = pb[0]; b.u[1] = pb[2];
      acc = __builtin_amdgcn_wmma_f32_16x16x32_bf16(
          false, a.v, false, b.v, (short)0, acc, false, false);
    }
    int col = ct * 16 + lcol;
    float bv = bias ? bias[col] : 0.f;
#pragma unroll
    for (int v = 0; v < 8; ++v) {
      int row = rt * 16 + v + 8 * half;
      float x = acc[v] + bv;
      if (act == 1) {
        const float kAlpha = 1.6732632423543772f, kScale = 1.0507009873554805f;
        x = (x > 0.f) ? kScale * x : kScale * kAlpha * (__expf(x) - 1.f);
      }
      C[(size_t)row * N + col] = x;
    }
  }
}

// ---------- batch norm over batch axis (rows = 256) ----------
__global__ void batchnorm_kernel(const float* __restrict__ X, int in_stride,
                                 const float* __restrict__ gamma, const float* __restrict__ beta,
                                 unsigned short* __restrict__ out_bf,
                                 float* __restrict__ out_f32, int out_stride, int C) {
  int c = blockIdx.x * blockDim.x + threadIdx.x;
  if (c >= C) return;
  float s = 0.f, s2 = 0.f;
  for (int r = 0; r < BB; ++r) {
    float x = X[(size_t)r * in_stride + c];
    s += x; s2 += x * x;
  }
  float mean = s * (1.f / BB);
  float var  = s2 * (1.f / BB) - mean * mean;
  float inv  = rsqrtf(var + 1e-3f) * gamma[c];
  float bet  = beta[c];
  for (int r = 0; r < BB; ++r) {
    float y = (X[(size_t)r * in_stride + c] - mean) * inv + bet;
    if (out_bf)  out_bf[(size_t)r * out_stride + c] = f2bf(y);
    if (out_f32) out_f32[(size_t)r * out_stride + c] = y;
  }
}

// ---------- rep = concat(r1, r2, r3) from last-timestep hidden ----------
__global__ void rep_extract_kernel(const unsigned short* __restrict__ Xlast, // [768][512]
                                   float* __restrict__ rep) {                // [256][1536]
  int idx = blockIdx.x * blockDim.x + threadIdx.x;
  if (idx >= BB * 3 * HH) return;
  int c = idx % HH;
  int text = (idx / HH) % 3;
  int b = idx / (3 * HH);
  rep[idx] = bf2f(Xlast[((size_t)text * BB + b) * HH + c]);
}

// ---------- tiny final FC: [256][102] @ [102][4] + b ----------
__global__ void fc3_kernel(const float* __restrict__ X, const float* __restrict__ W,
                           const float* __restrict__ b, float* __restrict__ out) {
  int idx = blockIdx.x * blockDim.x + threadIdx.x;
  if (idx >= BB * 4) return;
  int o = idx & 3, r = idx >> 2;
  float s = b[o];
  for (int k = 0; k < 102; ++k) s += X[(size_t)r * 102 + k] * W[k * 4 + o];
  out[idx] = s;
}

extern "C" void kernel_launch(void* const* d_in, const int* in_sizes, int n_in,
                              void* d_out, int out_size, void* d_ws, size_t ws_size,
                              hipStream_t stream) {
  (void)in_sizes; (void)n_in; (void)out_size; (void)ws_size;
  const int*   t1     = (const int*)d_in[0];
  const int*   t2     = (const int*)d_in[1];
  const int*   t3     = (const int*)d_in[2];
  const float* emb    = (const float*)d_in[3];
  const float* lstm_W = (const float*)d_in[4];
  const float* lstm_b = (const float*)d_in[5];
  const float* bn1_g  = (const float*)d_in[6];
  const float* bn1_b  = (const float*)d_in[7];
  const float* W1     = (const float*)d_in[8];
  const float* b1     = (const float*)d_in[9];
  const float* bn2_g  = (const float*)d_in[10];
  const float* bn2_b  = (const float*)d_in[11];
  const float* W2     = (const float*)d_in[12];
  const float* b2     = (const float*)d_in[13];
  const float* bn3_g  = (const float*)d_in[14];
  const float* bn3_b  = (const float*)d_in[15];
  const float* W3     = (const float*)d_in[16];
  const float* b3     = (const float*)d_in[17];
  float* out = (float*)d_out;

  char* ws = (char*)d_ws;
  size_t off = 0;
  auto alloc = [&](size_t bytes) -> char* {
    char* p = ws + off;
    off = (off + bytes + 255) & ~(size_t)255;
    return p;
  };
  int*            cnt  = (int*)alloc(256);
  unsigned short* Xa   = (unsigned short*)alloc((size_t)TT * RR * HH * 2);
  unsigned short* Xb   = (unsigned short*)alloc((size_t)TT * RR * HH * 2);
  unsigned short* WTl  = (unsigned short*)alloc((size_t)3 * ZZ * KK * 2);
  unsigned short* hA   = (unsigned short*)alloc((size_t)RR * HH * 2);
  unsigned short* hB   = (unsigned short*)alloc((size_t)RR * HH * 2);
  float*          cst  = (float*)alloc((size_t)RR * HH * 4);
  unsigned short* W1T  = (unsigned short*)alloc((size_t)1024 * 1536 * 2);
  unsigned short* W2T  = (unsigned short*)alloc((size_t)112 * 1024 * 2);
  float*          b2p  = (float*)alloc(112 * 4);
  float*          repf = (float*)alloc((size_t)BB * 1536 * 4);
  unsigned short* repb = (unsigned short*)alloc((size_t)BB * 1536 * 2);
  float*          f1f  = (float*)alloc((size_t)BB * 1024 * 4);
  unsigned short* f1b  = (unsigned short*)alloc((size_t)BB * 1024 * 2);
  float*          f2f  = (float*)alloc((size_t)BB * 112 * 4);
  float*          bn3f = (float*)alloc((size_t)BB * 102 * 4);

  // weight casts + embedding gather
  cast_lstm_w_kernel<<<(3 * ZZ * KK + 255) / 256, 256, 0, stream>>>(lstm_W, WTl);
  cast_w1_kernel<<<(1024 * 1536 + 255) / 256, 256, 0, stream>>>(W1, W1T);
  cast_w2_kernel<<<(112 * 1024 + 255) / 256, 256, 0, stream>>>(W2, b2, W2T, b2p);
  embed_gather_kernel<<<(TT * RR * (HH / 8) + 255) / 256, 256, 0, stream>>>(t1, t2, t3, emb, Xa);

  // 3 stacked LSTM layers, ping-pong sequence buffers
  const size_t lstm_lds = (size_t)64 * LDSTR * 2;   // 132096 B weight slice in LDS
  unsigned short* Xin = Xa;
  unsigned short* Xout = Xb;
  for (int l = 0; l < 3; ++l) {
    reset_barrier_kernel<<<1, 64, 0, stream>>>(cnt);
    lstm_layer_kernel<<<NWG, THREADS, lstm_lds, stream>>>(
        WTl + (size_t)l * ZZ * KK, lstm_b + (size_t)l * ZZ, Xin, Xout, hA, hB, cst, cnt);
    unsigned short* tmp = Xin; Xin = Xout; Xout = tmp;
  }
  // final layer output now in Xin

  // head
  rep_extract_kernel<<<(BB * 3 * HH + 255) / 256, 256, 0, stream>>>(
      Xin + (size_t)(TT - 1) * RR * HH, repf);
  batchnorm_kernel<<<(1536 + 255) / 256, 256, 0, stream>>>(
      repf, 1536, bn1_g, bn1_b, repb, nullptr, 1536, 1536);
  wmma_gemm_kernel<<<128, 256, 0, stream>>>(repb, W1T, b1, f1f, 256, 1024, 1536, 1);
  batchnorm_kernel<<<(1024 + 255) / 256, 256, 0, stream>>>(
      f1f, 1024, bn2_g, bn2_b, f1b, nullptr, 1024, 1024);
  wmma_gemm_kernel<<<16, 256, 0, stream>>>(f1b, W2T, b2p, f2f, 256, 112, 1024, 1);
  batchnorm_kernel<<<1, 128, 0, stream>>>(f2f, 112, bn3_g, bn3_b, nullptr, bn3f, 102, 102);
  fc3_kernel<<<(BB * 4 + 255) / 256, 256, 0, stream>>>(bn3f, W3, b3, out);
}
